// GravityGAE_8074538516900
// MI455X (gfx1250) — matrix operand order repaired and checked by reading, not compiled, not run on hardware
//
#include <hip/hip_runtime.h>

#define EPSILON 0.01f
#define N_ROWS 8192
#define IN_DIM 512
#define HID 64
#define ZP 48      // z padded cols (33 -> 48, 3 WMMA n-tiles)
#define ZD 32      // embedding dim used for distances
#define LN2F 0.6931471805599453f

typedef __attribute__((ext_vector_type(16))) __bf16 v16bf;
typedef __attribute__((ext_vector_type(8)))  float  v8f;
typedef __attribute__((ext_vector_type(2)))  float  v2f;

// dist >= EPSILON always, so skip libm's denormal-guarded log: raw v_log_f32.
__device__ __forceinline__ float fast_log(float x) {
  return LN2F * __builtin_amdgcn_logf(x);
}

// ---------------------------------------------------------------------------
// fp32 -> bf16 conversion helpers
// ---------------------------------------------------------------------------
__device__ __forceinline__ v16bf cvt16(float4 f0, float4 f1, float4 f2, float4 f3) {
  v16bf a;
  a[0]  = (__bf16)f0.x; a[1]  = (__bf16)f0.y; a[2]  = (__bf16)f0.z; a[3]  = (__bf16)f0.w;
  a[4]  = (__bf16)f1.x; a[5]  = (__bf16)f1.y; a[6]  = (__bf16)f1.z; a[7]  = (__bf16)f1.w;
  a[8]  = (__bf16)f2.x; a[9]  = (__bf16)f2.y; a[10] = (__bf16)f2.z; a[11] = (__bf16)f2.w;
  a[12] = (__bf16)f3.x; a[13] = (__bf16)f3.y; a[14] = (__bf16)f3.z; a[15] = (__bf16)f3.w;
  return a;
}

__global__ void cvt_bf16_kernel(const float* __restrict__ in, __bf16* __restrict__ out, int n) {
  int i = blockIdx.x * blockDim.x + threadIdx.x;
  if (i < n) out[i] = (__bf16)in[i];
}

// ---------------------------------------------------------------------------
// bf16 WMMA GEMM:  Out[M x NTILES*16] = op(A_f32[M x K] @ B_bf16[K x NTILES*16])
// A converted fp32->bf16 on the fly (saves an HBM staging pass for adj).
// One wave computes a 16-row x (NTILES*16)-col strip; 8 waves / block.
// A-frag layout (16b A 16x32): lanes 0-15: M=lane, K = kk+{0..7,16..23};
//                              lanes 16-31: M=lane-16, K = kk+{8..15,24..31}.
// B-frag layout (16b B 32x16): lane holds col N=lane&15, K same as A-frag.
// ---------------------------------------------------------------------------
template <int NTILES, bool RELU, bool OUT_F32>
__global__ __launch_bounds__(256) void gemm_bf16_kernel(
    const float* __restrict__ A, int lda,
    const __bf16* __restrict__ B, int ldb,
    int K, void* __restrict__ Out, int ldo)
{
  const int lane = threadIdx.x & 31;
  const int wave = threadIdx.x >> 5;
  const int mrow = (blockIdx.x * 8 + wave) * 16;

  v8f acc[NTILES];
  const v8f vzero = {0.f, 0.f, 0.f, 0.f, 0.f, 0.f, 0.f, 0.f};
#pragma unroll
  for (int t = 0; t < NTILES; ++t) acc[t] = vzero;

  const int m    = lane & 15;
  const int koff = (lane >> 4) << 3;                 // 0 or 8
  const float* __restrict__ arow = A + (size_t)(mrow + m) * lda;

  for (int kk = 0; kk < K; kk += 32) {
    // Prefetch the next adj chunk of this row into cache (global_prefetch_b8).
    __builtin_prefetch(arow + kk + 128, 0, 1);

    const float4 f0 = *(const float4*)(arow + kk + koff);
    const float4 f1 = *(const float4*)(arow + kk + koff + 4);
    const float4 f2 = *(const float4*)(arow + kk + koff + 16);
    const float4 f3 = *(const float4*)(arow + kk + koff + 20);
    const v16bf a = cvt16(f0, f1, f2, f3);

    const __bf16* __restrict__ brow = B + (size_t)(kk + koff) * ldb + m;
#pragma unroll
    for (int t = 0; t < NTILES; ++t) {
      v16bf b;
#pragma unroll
      for (int s = 0; s < 16; ++s)
        b[s] = brow[(size_t)(s < 8 ? s : s + 8) * ldb + t * 16];
      acc[t] = __builtin_amdgcn_wmma_f32_16x16x32_bf16(
          false, a, false, b, (short)0, acc[t], false, false);
    }
  }

  // Epilogue: C/D layout -> row = mrow + r + (lane>=16 ? 8 : 0), col = t*16 + (lane&15)
  const int row_lo = mrow + ((lane >> 4) << 3);
#pragma unroll
  for (int r = 0; r < 8; ++r) {
    const int row = row_lo + r;
#pragma unroll
    for (int t = 0; t < NTILES; ++t) {
      float v = acc[t][r];
      if (RELU) v = fmaxf(v, 0.f);
      const int col = t * 16 + (lane & 15);
      if (OUT_F32) ((float*)Out)[(size_t)row * ldo + col] = v;
      else         ((__bf16*)Out)[(size_t)row * ldo + col] = (__bf16)v;
    }
  }
}

// ---------------------------------------------------------------------------
// hW2 = h(bf16, 8192x64) @ W2(f32, 64x33), padded to 48 cols, stored bf16.
// ---------------------------------------------------------------------------
__global__ void hw2_kernel(const __bf16* __restrict__ h, const float* __restrict__ W2,
                           __bf16* __restrict__ out) {
  const int idx = blockIdx.x * blockDim.x + threadIdx.x;   // 8192*48 threads
  const int row = idx / ZP;
  const int col = idx - row * ZP;
  float s = 0.f;
  if (col < 33) {
    const __bf16* __restrict__ hr = h + (size_t)row * HID;
#pragma unroll 16
    for (int k = 0; k < HID; ++k) s += (float)hr[k] * W2[k * 33 + col];
  }
  out[idx] = (__bf16)s;
}

// ---------------------------------------------------------------------------
// x1[i] = sum_{n<32} z[i][n]^2 ; mcol[i] = z[i][32]
// ---------------------------------------------------------------------------
__global__ void rowstats_kernel(const float* __restrict__ z,
                                float* __restrict__ x1, float* __restrict__ mcol) {
  const int i = blockIdx.x * blockDim.x + threadIdx.x;
  if (i >= N_ROWS) return;
  const float* __restrict__ zr = z + (size_t)i * ZP;
  float s = 0.f;
#pragma unroll
  for (int n = 0; n < ZD; ++n) s += zr[n] * zr[n];
  x1[i]   = s;
  mcol[i] = zr[ZD];
}

// ---------------------------------------------------------------------------
// out[i][j] = m[j] - log(x1[i] + x1[j] - 2 * dot(z_i, z_j) + eps)
// One wave per 16x16 tile; dot via 8x V_WMMA_F32_16X16X4_F32 (fp32 keeps the
// diagonal cancellation x1_i - z_i.z_i exact to fp32 rounding vs eps=0.01).
//
// The 16 i-rows of z (3 KB) are shared by all 8 waves of the block: staged
// once into LDS with the CDNA5 async path (global_load_async_to_lds_b128 +
// s_wait_asynccnt), then A-fragments come from ds_load instead of 8 redundant
// global streams. j-rows differ per wave and stay on the global path.
// ---------------------------------------------------------------------------
__global__ __launch_bounds__(256) void pairwise_kernel(
    const float* __restrict__ z, const float* __restrict__ x1,
    const float* __restrict__ mcol, float* __restrict__ out)
{
  __shared__ float ztile[16 * ZP];                   // 3 KB: rows i0..i0+15 of z

  const int tid  = threadIdx.x;
  const int lane = threadIdx.x & 31;
  const int wave = threadIdx.x >> 5;
  const int i0 = blockIdx.y * 16;
  const int j0 = (blockIdx.x * 8 + wave) * 16;

  // Async-stage the i-tile: 16*48 floats = 768 dwords = 192 x b128.
  if (tid < (16 * ZP) / 4) {
    const float* gsrc = z + (size_t)i0 * ZP + tid * 4;
    unsigned lds_off = (unsigned)(uintptr_t)(&ztile[tid * 4]);  // LDS aperture: low 32 bits
    asm volatile("global_load_async_to_lds_b128 %0, %1, off"
                 :: "v"(lds_off), "v"(gsrc)
                 : "memory");
  }
  asm volatile("s_wait_asynccnt 0x0" ::: "memory");
  __syncthreads();

  const int r     = lane & 15;
  const int phase = (lane >> 4) << 1;                // 0 or 2
  const float* zi = &ztile[r * ZP];                  // LDS
  const float* __restrict__ zj = z + (size_t)(j0 + r) * ZP;

  v8f c = {0.f, 0.f, 0.f, 0.f, 0.f, 0.f, 0.f, 0.f};
#pragma unroll
  for (int s = 0; s < 8; ++s) {
    const float2 af = *(const float2*)(zi + 4 * s + phase);   // ds_load_b64
    const float2 bf = *(const float2*)(zj + 4 * s + phase);   // global_load_b64
    v2f a; a[0] = af.x; a[1] = af.y;
    v2f b; b[0] = bf.x; b[1] = bf.y;
    c = __builtin_amdgcn_wmma_f32_16x16x4_f32(
        false, a, false, b, (short)0, c, false, false);
  }

  const float x1j = x1[j0 + r];
  const float mj  = mcol[j0 + r];
  const float base = x1j + EPSILON;
  const int row_lo = i0 + ((lane >> 4) << 3);
#pragma unroll
  for (int rr = 0; rr < 8; ++rr) {
    const int row = row_lo + rr;
    const float dist = fmaf(-2.0f, c[rr], x1[row] + base);
    out[(size_t)row * N_ROWS + (j0 + r)] = mj - fast_log(dist);
  }
}

// ---------------------------------------------------------------------------
extern "C" void kernel_launch(void* const* d_in, const int* in_sizes, int n_in,
                              void* d_out, int out_size, void* d_ws, size_t ws_size,
                              hipStream_t stream) {
  const float* x   = (const float*)d_in[0];   // 8192 x 512
  const float* adj = (const float*)d_in[1];   // 8192 x 8192
  const float* W1  = (const float*)d_in[2];   // 512 x 64
  const float* W2  = (const float*)d_in[3];   // 64 x 33
  float* out = (float*)d_out;                  // 8192 x 8192

  // Workspace carve-up (256B aligned)
  uintptr_t p = (uintptr_t)d_ws;
  auto take = [&](size_t bytes) {
    p = (p + 255) & ~(uintptr_t)255;
    uintptr_t r = p; p += bytes; return (void*)r;
  };
  __bf16* W1b   = (__bf16*)take((size_t)IN_DIM * HID * 2);   // 64 KB
  __bf16* xW1b  = (__bf16*)take((size_t)N_ROWS * HID * 2);   // 1 MB
  __bf16* hb    = (__bf16*)take((size_t)N_ROWS * HID * 2);   // 1 MB
  __bf16* hW2b  = (__bf16*)take((size_t)N_ROWS * ZP * 2);    // 768 KB
  float*  zbuf  = (float*) take((size_t)N_ROWS * ZP * 4);    // 1.5 MB
  float*  x1    = (float*) take((size_t)N_ROWS * 4);
  float*  mcol  = (float*) take((size_t)N_ROWS * 4);
  (void)ws_size; (void)in_sizes; (void)n_in; (void)out_size;

  // 1) W1 -> bf16
  cvt_bf16_kernel<<<(IN_DIM * HID + 255) / 256, 256, 0, stream>>>(W1, W1b, IN_DIM * HID);

  // 2) xW1 = x @ W1  (bf16 out, row-major 8192x64)
  gemm_bf16_kernel<4, false, false><<<N_ROWS / 128, 256, 0, stream>>>(
      x, IN_DIM, W1b, HID, IN_DIM, xW1b, HID);

  // 3) h = relu(adj @ xW1)  (bf16 out)
  gemm_bf16_kernel<4, true, false><<<N_ROWS / 128, 256, 0, stream>>>(
      adj, N_ROWS, xW1b, HID, N_ROWS, hb, HID);

  // 4) hW2 = h @ W2, padded to 48 cols, bf16
  hw2_kernel<<<(N_ROWS * ZP) / 256, 256, 0, stream>>>(hb, W2, hW2b);

  // 5) z = adj @ hW2  (fp32 out, ld=48)
  gemm_bf16_kernel<3, false, true><<<N_ROWS / 128, 256, 0, stream>>>(
      adj, N_ROWS, hW2b, ZP, N_ROWS, zbuf, ZP);

  // 6) x1, m
  rowstats_kernel<<<N_ROWS / 256, 256, 0, stream>>>(zbuf, x1, mcol);

  // 7) out = m.T - log(x1 + x1.T - 2 z z^T + eps)
  pairwise_kernel<<<dim3(N_ROWS / 128, N_ROWS / 16), 256, 0, stream>>>(zbuf, x1, mcol, out);
}
